// AttnHead_61658550502133
// MI455X (gfx1250) — matrix operand order repaired and checked by reading, not compiled, not run on hardware
//
#include <hip/hip_runtime.h>
#include <hip/hip_bf16.h>

typedef __attribute__((ext_vector_type(16))) _Float16 v16h;
typedef __attribute__((ext_vector_type(8)))  _Float16 v8h;
typedef __attribute__((ext_vector_type(8)))  float    v8f;

#define IN_DIM  256
#define OUT_DIM 128

// ---- order-preserving float <-> uint encoding (for atomic segment max) ----
__device__ __forceinline__ unsigned enc_f32(float x) {
  unsigned b = __float_as_uint(x);
  return (b & 0x80000000u) ? ~b : (b | 0x80000000u);
}
__device__ __forceinline__ float dec_f32(unsigned k) {
  unsigned b = (k & 0x80000000u) ? (k ^ 0x80000000u) : ~k;
  return __uint_as_float(b);
}

// ============================================================================
// K0: repack W [256 x 128] f32 -> f16 in WMMA B-operand layout.
// B (32x16 per K-step) layout: lanes 0-15 hold K=0..7 (halves 0..7), K=16..23
// (halves 8..15); lanes 16-31: +8 on the K base; N = lane & 15.
// Linearized: Wp[((ntile*8 + kk)*32 + lane)*16 + i]
// ============================================================================
__global__ void pack_w(const float* __restrict__ W, _Float16* __restrict__ Wp) {
  int idx = blockIdx.x * 256 + threadIdx.x;          // 8*8*32*16 = 32768 total
  if (idx >= 8 * 8 * 32 * 16) return;
  int i    = idx & 15;
  int lane = (idx >> 4) & 31;
  int kk   = (idx >> 9) & 7;
  int nt   = idx >> 12;
  int n = nt * 16 + (lane & 15);
  int k = kk * 32 + ((i >> 3) << 4) + ((lane >> 4) << 3) + (i & 7);
  Wp[idx] = (_Float16)W[k * OUT_DIM + n];
}

// ============================================================================
// K1: seq = feat @ W via v_wmma_f32_16x16x32_f16.
// Block = 256 threads = 8 waves; block computes rows [16) x all 128 cols.
// A slab (16 x 256) staged in LDS as f16; row stride 264 halves (528B, 16B
// aligned) so per-lane A fragment = two 16B ds loads.
// A layout (ISA table): lanes 0-15: halves 0..7 = K 0..7, halves 8..15 = K16..23;
// lanes 16-31: +8 on the K base.
// ============================================================================
__global__ void __launch_bounds__(256) gemm_wmma(const float* __restrict__ feat,
                                                 const _Float16* __restrict__ Wp,
                                                 float* __restrict__ seq, int N) {
  __shared__ _Float16 Ash[16][264];
  const int tid  = threadIdx.x;
  const int row0 = blockIdx.x * 16;

  // cooperative stage: each thread loads 16 consecutive f32, converts to f16
  {
    int r    = tid >> 4;          // 0..15
    int c0   = (tid & 15) * 16;   // 0,16,...,240
    int grow = row0 + r;
    if (grow < N) {
      const float4* src = (const float4*)(feat + (size_t)grow * IN_DIM + c0);
      #pragma unroll
      for (int j = 0; j < 4; ++j) {
        float4 v = src[j];
        Ash[r][c0 + 4 * j + 0] = (_Float16)v.x;
        Ash[r][c0 + 4 * j + 1] = (_Float16)v.y;
        Ash[r][c0 + 4 * j + 2] = (_Float16)v.z;
        Ash[r][c0 + 4 * j + 3] = (_Float16)v.w;
      }
    } else {
      #pragma unroll
      for (int j = 0; j < 16; ++j) Ash[r][c0 + j] = (_Float16)0.0f;
    }
  }
  __syncthreads();

  const int wave = tid >> 5;            // 0..7  -> 16-col stripe
  const int lane = tid & 31;
  const int m    = lane & 15;
  const int kb   = (lane >> 4) << 3;    // 0 or 8 (A layout K base)

  v8f c = {};
  #pragma unroll
  for (int kk = 0; kk < 8; ++kk) {
    const v8h a_lo = *(const v8h*)&Ash[m][kk * 32 + kb];
    const v8h a_hi = *(const v8h*)&Ash[m][kk * 32 + kb + 16];
    v16h a = __builtin_shufflevector(a_lo, a_hi,
                                     0, 1, 2, 3, 4, 5, 6, 7,
                                     8, 9, 10, 11, 12, 13, 14, 15);
    v16h b = *(const v16h*)(Wp + ((((size_t)wave * 8 + kk) * 32 + lane) << 4));
    c = __builtin_amdgcn_wmma_f32_16x16x32_f16(false, a, false, b,
                                               (short)0, c, false, false);
  }

  // C/D layout: lane n = lane&15 (col), VGPR r -> row r + 8*(lane>=16)
  const int n  = wave * 16 + m;
  const int mb = (lane >> 4) << 3;
  #pragma unroll
  for (int r = 0; r < 8; ++r) {
    int grow = row0 + mb + r;
    if (grow < N) seq[(size_t)grow * OUT_DIM + n] = c[r];
  }
}

// ============================================================================
// K2: f1 = seq@a_l + b_l, f2 = seq@a_r + b_r (wave per node), and init
// segment-max (encoded) and segment-sum buffers.
// ============================================================================
__global__ void __launch_bounds__(256) f1f2_init(const float* __restrict__ seq,
    const float* __restrict__ al, const float* __restrict__ bl,
    const float* __restrict__ ar, const float* __restrict__ br,
    float* __restrict__ f1, float* __restrict__ f2,
    unsigned* __restrict__ m_enc, float* __restrict__ s, int N) {
  int wave = threadIdx.x >> 5, lane = threadIdx.x & 31;
  int node = blockIdx.x * 8 + wave;
  if (node >= N) return;
  float4 v = ((const float4*)(seq + (size_t)node * OUT_DIM))[lane];
  float4 a = ((const float4*)al)[lane];
  float4 b = ((const float4*)ar)[lane];
  float p1 = v.x * a.x + v.y * a.y + v.z * a.z + v.w * a.w;
  float p2 = v.x * b.x + v.y * b.y + v.z * b.z + v.w * b.w;
  #pragma unroll
  for (int off = 16; off; off >>= 1) {
    p1 += __shfl_xor(p1, off, 32);
    p2 += __shfl_xor(p2, off, 32);
  }
  if (lane == 0) {
    f1[node]    = p1 + bl[0];
    f2[node]    = p2 + br[0];
    m_enc[node] = 0u;          // decodes below any finite logit
    s[node]     = 0.0f;
  }
}

// K3: zero the output accumulator (float4 granularity).
__global__ void zero_out(float4* __restrict__ out, int count4) {
  int i = blockIdx.x * 256 + threadIdx.x;
  if (i < count4) out[i] = make_float4(0.f, 0.f, 0.f, 0.f);
}

// K4: per-edge leaky-relu logit + atomic segment max (encoded uint).
__global__ void edge_logit_max(const int* __restrict__ row, const int* __restrict__ col,
                               const float* __restrict__ f1, const float* __restrict__ f2,
                               float* __restrict__ logit, unsigned* __restrict__ m_enc, int E) {
  int e = blockIdx.x * 256 + threadIdx.x;
  if (e >= E) return;
  int r = row[e];
  float x = f1[r] + f2[col[e]];
  float l = x > 0.0f ? x : 0.2f * x;
  logit[e] = l;
  atomicMax(m_enc + r, enc_f32(l));
}

// K5: e = exp(logit - m[row]) (in place over logit buffer) + atomic segment sum.
__global__ void edge_exp_sum(const int* __restrict__ row, float* __restrict__ ebuf,
                             const unsigned* __restrict__ m_enc, float* __restrict__ s, int E) {
  int e = blockIdx.x * 256 + threadIdx.x;
  if (e >= E) return;
  int r = row[e];
  float ev = __expf(ebuf[e] - dec_f32(m_enc[r]));
  ebuf[e] = ev;
  atomicAdd(s + r, ev);
}

// K6: SpMM — wave per edge, 4 cols/lane; seq rows are L2-resident (51MB << 192MB).
__global__ void __launch_bounds__(256) spmm(const int* __restrict__ row, const int* __restrict__ col,
                                            const float* __restrict__ ebuf, const float* __restrict__ s,
                                            const float* __restrict__ seq, float* __restrict__ out, int E) {
  int wave = threadIdx.x >> 5, lane = threadIdx.x & 31;
  int e = blockIdx.x * 8 + wave;
  if (e >= E) return;
  int r = row[e], cdx = col[e];
  float coef = ebuf[e] / s[r];
  float4 v = ((const float4*)(seq + (size_t)cdx * OUT_DIM))[lane];
  float* dst = out + (size_t)r * OUT_DIM + lane * 4;
  atomicAdd(dst + 0, coef * v.x);
  atomicAdd(dst + 1, coef * v.y);
  atomicAdd(dst + 2, coef * v.z);
  atomicAdd(dst + 3, coef * v.w);
}

// K7: out = elu(out + bias), in place.
__global__ void elu_bias(float* __restrict__ out, const float* __restrict__ bias, int total) {
  int i = blockIdx.x * 256 + threadIdx.x;
  if (i >= total) return;
  float x = out[i] + bias[i & (OUT_DIM - 1)];
  out[i] = x > 0.0f ? x : (__expf(x) - 1.0f);
}

extern "C" void kernel_launch(void* const* d_in, const int* in_sizes, int n_in,
                              void* d_out, int out_size, void* d_ws, size_t ws_size,
                              hipStream_t stream) {
  const float* feat = (const float*)d_in[0];
  const float* W    = (const float*)d_in[1];
  const float* a_l  = (const float*)d_in[2];
  const float* b_l  = (const float*)d_in[3];
  const float* a_r  = (const float*)d_in[4];
  const float* b_r  = (const float*)d_in[5];
  const float* bias = (const float*)d_in[6];
  const int*   row  = (const int*)d_in[7];
  const int*   col  = (const int*)d_in[8];

  const int N = in_sizes[0] / IN_DIM;
  const int E = in_sizes[7];
  float* out = (float*)d_out;

  // workspace layout (256B aligned chunks)
  char* ws = (char*)d_ws;
  size_t off = 0;
  auto alloc = [&](size_t bytes) -> void* {
    void* p = ws + off;
    off += (bytes + 255) & ~(size_t)255;
    return p;
  };
  float*     seq   = (float*)alloc((size_t)N * OUT_DIM * sizeof(float));
  _Float16*  Wp    = (_Float16*)alloc((size_t)8 * 8 * 32 * 16 * sizeof(_Float16));
  float*     f1    = (float*)alloc((size_t)N * sizeof(float));
  float*     f2    = (float*)alloc((size_t)N * sizeof(float));
  unsigned*  m_enc = (unsigned*)alloc((size_t)N * sizeof(unsigned));
  float*     s     = (float*)alloc((size_t)N * sizeof(float));
  float*     ebuf  = (float*)alloc((size_t)E * sizeof(float));

  pack_w<<<dim3((8 * 8 * 32 * 16 + 255) / 256), dim3(256), 0, stream>>>(W, Wp);
  gemm_wmma<<<dim3((N + 15) / 16), dim3(256), 0, stream>>>(feat, Wp, seq, N);
  f1f2_init<<<dim3((N + 7) / 8), dim3(256), 0, stream>>>(seq, a_l, b_l, a_r, b_r,
                                                         f1, f2, m_enc, s, N);
  zero_out<<<dim3((N * (OUT_DIM / 4) + 255) / 256), dim3(256), 0, stream>>>((float4*)out,
                                                                            N * (OUT_DIM / 4));
  edge_logit_max<<<dim3((E + 255) / 256), dim3(256), 0, stream>>>(row, col, f1, f2,
                                                                  ebuf, m_enc, E);
  edge_exp_sum<<<dim3((E + 255) / 256), dim3(256), 0, stream>>>(row, ebuf, m_enc, s, E);
  spmm<<<dim3((E + 7) / 8), dim3(256), 0, stream>>>(row, col, ebuf, s, seq, out, E);
  elu_bias<<<dim3((N * OUT_DIM + 255) / 256), dim3(256), 0, stream>>>(out, bias, N * OUT_DIM);
}
